// ContrastivePerceptualLoss_10445360464140
// MI455X (gfx1250) — matrix operand order repaired and checked by reading, not compiled, not run on hardware
//
#include <hip/hip_runtime.h>
#include <hip/hip_bf16.h>

// ---------------------------------------------------------------------------
// CDNA5 (gfx1250) WMMA implementation of the VGG contrastive perceptual loss.
// Convs run as implicit GEMM on v_wmma_f32_16x16x32_f16 (f32 accumulate),
// with double-buffered async global->LDS staging (ASYNCcnt) to overlap
// the next tile's loads with the current tile's WMMAs.
// ---------------------------------------------------------------------------

typedef __attribute__((ext_vector_type(16))) _Float16 v16h;
typedef __attribute__((ext_vector_type(8)))  _Float16 v8h;
typedef __attribute__((ext_vector_type(8)))  float    v8f;
typedef int i32x4 __attribute__((__vector_size__(16)));   // matches builtin param

union V16 { v16h v; v8h h[2]; };

// ---- async global->LDS (gfx1250) with synchronous fallback ----------------
#if defined(__has_builtin)
#if __has_builtin(__builtin_amdgcn_global_load_async_to_lds_b128) && \
    __has_builtin(__builtin_amdgcn_s_wait_asynccnt)
#define USE_ASYNC_LDS 1
#endif
#endif
#ifndef USE_ASYNC_LDS
#define USE_ASYNC_LDS 0
#endif

// Copy 8 f16 (16B) from global to LDS.
__device__ __forceinline__ void g2l_16(const _Float16* src, _Float16* dst) {
#if USE_ASYNC_LDS
    __builtin_amdgcn_global_load_async_to_lds_b128(
        (__attribute__((address_space(1))) i32x4*)src,
        (__attribute__((address_space(3))) i32x4*)dst, 0, 0);
#else
    *reinterpret_cast<float4*>(dst) = *reinterpret_cast<const float4*>(src);
#endif
}

__device__ __forceinline__ void async_fence() {
#if USE_ASYNC_LDS
    __builtin_amdgcn_s_wait_asynccnt(0);
#endif
}

__device__ __forceinline__ void zero_16(_Float16* dst) {
    float4 z = {0.f, 0.f, 0.f, 0.f};
    *reinterpret_cast<float4*>(dst) = z;
}

// ---------------------------------------------------------------------------
// Weight transform: f32 OIHW -> f16 [OC][KP], K index = (ky*3+kx)*IC + ic.
// ---------------------------------------------------------------------------
__global__ void cvt_weights(const float* __restrict__ w, _Float16* __restrict__ dst,
                            int OC, int IC, int KP) {
    int i = blockIdx.x * blockDim.x + threadIdx.x;
    int total = OC * KP;
    if (i >= total) return;
    int oc = i / KP;
    int k  = i - oc * KP;
    _Float16 val = (_Float16)0.f;
    if (k < 9 * IC) {
        int tap = k / IC;
        int ic  = k - tap * IC;
        int ky  = tap / 3, kx = tap - ky * 3;
        val = (_Float16)w[((oc * IC + ic) * 3 + ky) * 3 + kx];
    }
    dst[i] = val;
}

// ---------------------------------------------------------------------------
// conv0: IC=3 (NCHW f32 source), OC=64, K padded 27->32, single K-step.
// Block 128 threads (4 waves), tile 64 pixels x 64 channels. ~0.5% of FLOPs.
// ---------------------------------------------------------------------------
__global__ void conv0_wmma(const float* __restrict__ in, const _Float16* __restrict__ wt,
                           const float* __restrict__ bias, _Float16* __restrict__ out) {
    const int H = 224, W = 224, OC = 64, HW = H * W;
    __shared__ alignas(16) _Float16 As[64 * 32];
    __shared__ alignas(16) _Float16 Bs[64 * 32];
    __shared__ alignas(16) _Float16 Cs[64 * 64];

    int b  = blockIdx.z;
    int p0 = blockIdx.x * 64;
    int t  = threadIdx.x;
    int wv = t >> 5, lane = t & 31;

    // Gather A: 64 rows x 32 K (27 valid) from NCHW f32.
    {
        int row = t >> 1;
        int p = p0 + row;
        int y = p / W, x = p - y * W;
        int off = (t & 1) * 16;
        for (int e = 0; e < 16; ++e) {
            int k = off + e;
            _Float16 val = (_Float16)0.f;
            if (k < 27) {
                int tap = k / 3, c = k - tap * 3;
                int ky = tap / 3, kx = tap - ky * 3;
                int iy = y + ky - 1, ix = x + kx - 1;
                if (iy >= 0 && iy < H && ix >= 0 && ix < W)
                    val = (_Float16)in[(((size_t)b * 3 + c) * H + iy) * W + ix];
            }
            As[row * 32 + k] = val;
        }
    }
    // B: padded weights [64][32]
    {
        int row = t >> 1, off = (t & 1) * 16;
        g2l_16(&wt[row * 32 + off],     &Bs[row * 32 + off]);
        g2l_16(&wt[row * 32 + off + 8], &Bs[row * 32 + off + 8]);
    }
    async_fence();
    __syncthreads();

    int kb  = (lane >> 4) * 8;
    int kb2 = (lane >> 4) * 16;
    const _Float16* ar = &As[(wv * 16 + (lane & 15)) * 32];
    V16 a;
    a.h[0] = *reinterpret_cast<const v8h*>(ar + kb);
    a.h[1] = *reinterpret_cast<const v8h*>(ar + 16 + kb);

    v8f acc[4] = {};
#pragma unroll
    for (int nt = 0; nt < 4; ++nt) {
        const _Float16* br = &Bs[(nt * 16 + (lane & 15)) * 32 + kb];
        V16 bb;
        bb.h[0] = *reinterpret_cast<const v8h*>(br);
        bb.h[1] = *reinterpret_cast<const v8h*>(br + 16);
        acc[nt] = __builtin_amdgcn_wmma_f32_16x16x32_f16(false, a.v, false, bb.v,
                                                         (short)0, acc[nt], false, false);
    }
    __syncthreads();

    // Stage C in LDS (immediate-offset b16 stores), then coalesced b128 out.
    {
        _Float16* cbase = &Cs[(size_t)(wv * 16 + ((lane >> 4) << 3)) * 64 + (lane & 15)];
#pragma unroll
        for (int nt = 0; nt < 4; ++nt) {
            float bv = bias[nt * 16 + (lane & 15)];
#pragma unroll
            for (int v = 0; v < 8; ++v) {
                float r = acc[nt][v] + bv;
                cbase[v * 64 + nt * 16] = (_Float16)(r > 0.f ? r : 0.f);
            }
        }
    }
    __syncthreads();
    {
        int row = t >> 1, off = (t & 1) * 32;  // 2 threads per 64-ch row
        _Float16* dst = out + ((size_t)b * HW + p0 + row) * OC + off;
        const _Float16* src = &Cs[row * 64 + off];
        *reinterpret_cast<float4*>(dst)      = *reinterpret_cast<const float4*>(src);
        *reinterpret_cast<float4*>(dst + 8)  = *reinterpret_cast<const float4*>(src + 8);
        *reinterpret_cast<float4*>(dst + 16) = *reinterpret_cast<const float4*>(src + 16);
        *reinterpret_cast<float4*>(dst + 24) = *reinterpret_cast<const float4*>(src + 24);
    }
}

// ---------------------------------------------------------------------------
// Generic conv (IC multiple of 32): NHWC f16 in/out, 3x3 SAME, bias+ReLU.
// Block 128 threads (4 waves). Tile: 128 pixels x 64 out-channels.
// Double-buffered LDS: next K-step's async loads overlap current WMMAs.
// grid = (ceil(H*W/128), OC/64, B).
// ---------------------------------------------------------------------------
template <int IC, int OC>
__global__ void conv_wmma(const _Float16* __restrict__ in, const _Float16* __restrict__ wt,
                          const float* __restrict__ bias, _Float16* __restrict__ out,
                          int H, int W) {
    constexpr int KT = 9 * IC;
    constexpr int KSTEPS = KT / 32;
    __shared__ alignas(16) _Float16 As[2][128 * 32];  // 2 x 8 KB
    __shared__ alignas(16) _Float16 Bs[2][64 * 32];   // 2 x 4 KB
    __shared__ alignas(16) _Float16 Cs[128 * 64];     // 16 KB

    int b   = blockIdx.z;
    int oc0 = blockIdx.y * 64;
    int p0  = blockIdx.x * 128;
    int HW  = H * W;
    int t   = threadIdx.x;
    int wv  = t >> 5, lane = t & 31;

    const _Float16* inB = in + (size_t)b * HW * IC;

    v8f acc[2][4] = {};

    int  pa  = p0 + t;                 // pixel row staged by this thread
    bool aok = pa < HW;
    int  pc  = aok ? pa : 0;
    int  py  = pc / W, px = pc - py * W;

    int brow = t >> 1;                 // B staging row
    int boff = (t & 1) * 16;

    auto stageA = [&](int ks, int bufI) {
        int k0  = ks * 32;
        int tap = k0 / IC;
        int icb = k0 - tap * IC;       // 32-chunk lies inside one (ky,kx) tap
        int ky  = tap / 3, kx = tap - ky * 3;
        int iy = py + ky - 1, ix = px + kx - 1;
        _Float16* dst = &As[bufI][t * 32];
        if (aok && iy >= 0 && iy < H && ix >= 0 && ix < W) {
            const _Float16* src = inB + ((size_t)iy * W + ix) * IC + icb;
            g2l_16(src,      dst);
            g2l_16(src + 8,  dst + 8);
            g2l_16(src + 16, dst + 16);
            g2l_16(src + 24, dst + 24);
        } else {
            zero_16(dst); zero_16(dst + 8); zero_16(dst + 16); zero_16(dst + 24);
        }
    };
    auto stageB = [&](int ks, int bufI) {
        const _Float16* src = &wt[(size_t)(oc0 + brow) * KT + ks * 32 + boff];
        g2l_16(src,     &Bs[bufI][brow * 32 + boff]);
        g2l_16(src + 8, &Bs[bufI][brow * 32 + boff + 8]);
    };

    // Prologue: stage K-step 0 into buffer 0.
    stageA(0, 0);
    stageB(0, 0);
    async_fence();
    __syncthreads();

    for (int ks = 0; ks < KSTEPS; ++ks) {
        int cur = ks & 1, nxt = cur ^ 1;
        // Kick off next tile's async loads; they run under the WMMAs below.
        if (ks + 1 < KSTEPS) {
            stageA(ks + 1, nxt);
            stageB(ks + 1, nxt);
        }

        // ---- WMMA: wave wv owns pixel rows [wv*32, wv*32+32) x 64 channels ----
        int kb  = (lane >> 4) * 8;
        int kb2 = (lane >> 4) * 16;
        const _Float16* ar0 = &As[cur][(wv * 32 + (lane & 15)) * 32];
        const _Float16* ar1 = ar0 + 16 * 32;
        V16 a0, a1;
        a0.h[0] = *reinterpret_cast<const v8h*>(ar0 + kb);
        a0.h[1] = *reinterpret_cast<const v8h*>(ar0 + 16 + kb);
        a1.h[0] = *reinterpret_cast<const v8h*>(ar1 + kb);
        a1.h[1] = *reinterpret_cast<const v8h*>(ar1 + 16 + kb);
#pragma unroll
        for (int nt = 0; nt < 4; ++nt) {
            const _Float16* br = &Bs[cur][(nt * 16 + (lane & 15)) * 32 + kb];
            V16 bb;
            bb.h[0] = *reinterpret_cast<const v8h*>(br);
            bb.h[1] = *reinterpret_cast<const v8h*>(br + 16);
            acc[0][nt] = __builtin_amdgcn_wmma_f32_16x16x32_f16(false, a0.v, false, bb.v,
                                                                (short)0, acc[0][nt], false, false);
            acc[1][nt] = __builtin_amdgcn_wmma_f32_16x16x32_f16(false, a1.v, false, bb.v,
                                                                (short)0, acc[1][nt], false, false);
        }
        // Wait for this iteration's async copies (overlapped with WMMAs above),
        // then rendezvous so the next iteration may consume the other buffer.
        async_fence();
        __syncthreads();
    }

    // ---- epilogue: bias+ReLU -> LDS stage -> coalesced NHWC b128 stores ----
    {
        float bv[4];
#pragma unroll
        for (int nt = 0; nt < 4; ++nt) bv[nt] = bias[oc0 + nt * 16 + (lane & 15)];
#pragma unroll
        for (int h = 0; h < 2; ++h) {
            _Float16* cbase =
                &Cs[(size_t)(wv * 32 + h * 16 + ((lane >> 4) << 3)) * 64 + (lane & 15)];
#pragma unroll
            for (int nt = 0; nt < 4; ++nt) {
#pragma unroll
                for (int v = 0; v < 8; ++v) {
                    float r = acc[h][nt][v] + bv[nt];
                    cbase[v * 64 + nt * 16] = (_Float16)(r > 0.f ? r : 0.f);
                }
            }
        }
    }
    __syncthreads();
    if (aok) {
        _Float16* dst = out + ((size_t)b * HW + pa) * OC + oc0;
        const _Float16* src = &Cs[t * 64];
#pragma unroll
        for (int q = 0; q < 4; ++q)
            *reinterpret_cast<float4*>(dst + q * 8) =
                *reinterpret_cast<const float4*>(src + q * 8);
    }
}

// ---------------------------------------------------------------------------
// 2x2 maxpool, NHWC f16.
// ---------------------------------------------------------------------------
__global__ void maxpool2(const _Float16* __restrict__ in, _Float16* __restrict__ out,
                         int H, int W, int C, int total) {
    int Wo = W >> 1, Ho = H >> 1;
    for (int i = blockIdx.x * blockDim.x + threadIdx.x; i < total;
         i += gridDim.x * blockDim.x) {
        int c = i % C;
        int r = i / C;
        int x = r % Wo; r /= Wo;
        int y = r % Ho;
        int b = r / Ho;
        const _Float16* base = in + (((size_t)b * H + 2 * y) * W + 2 * x) * C + c;
        float m = (float)base[0];
        m = fmaxf(m, (float)base[C]);
        m = fmaxf(m, (float)base[(size_t)W * C]);
        m = fmaxf(m, (float)base[(size_t)W * C + C]);
        out[i] = (_Float16)m;
    }
}

// ---------------------------------------------------------------------------
// 8x8 mean-pooled patches from NHWC [B,56,56,256] f16 -> f32 [B,256p,256c].
// Start indices clamped to [0,48] (jax dynamic_slice semantics).
// ---------------------------------------------------------------------------
__global__ void patch_mean(const _Float16* __restrict__ feat, const int* __restrict__ ii,
                           const int* __restrict__ jj, float* __restrict__ praw) {
    int p = blockIdx.x, b = blockIdx.y, c = threadIdx.x;
    int i = ii[p * 8 + b]; i = i < 0 ? 0 : (i > 48 ? 48 : i);
    int j = jj[p * 8 + b]; j = j < 0 ? 0 : (j > 48 ? 48 : j);
    float s = 0.f;
    for (int dy = 0; dy < 8; ++dy)
        for (int dx = 0; dx < 8; ++dx)
            s += (float)feat[(((size_t)b * 56 + i + dy) * 56 + j + dx) * 256 + c];
    praw[((size_t)b * 256 + p) * 256 + c] = s * (1.f / 64.f);
}

// ---------------------------------------------------------------------------
// L2 normalize each 256-d patch vector; output f16 for the WMMA similarity.
// ---------------------------------------------------------------------------
__global__ void l2norm_f16(const float* __restrict__ praw, _Float16* __restrict__ pf16) {
    int p = blockIdx.x, b = blockIdx.y, t = threadIdx.x;
    size_t base = ((size_t)b * 256 + p) * 256;
    float v = praw[base + t];
    float ss = v * v;
    for (int o = 16; o > 0; o >>= 1) ss += __shfl_xor(ss, o, 32);
    __shared__ float red[8];
    __shared__ float norm;
    if ((t & 31) == 0) red[t >> 5] = ss;
    __syncthreads();
    if (t == 0) {
        float s = 0.f;
        for (int i = 0; i < 8; ++i) s += red[i];
        norm = fmaxf(sqrtf(s), 1e-12f);
    }
    __syncthreads();
    pf16[base + t] = (_Float16)(v / norm);
}

// ---------------------------------------------------------------------------
// Similarity: S[b] = ip[b] (256x256) . tp[b]^T / TEMP via WMMA,
// double-buffered like the conv kernel.
// grid = (4 m-tiles, 4 n-tiles, 8 batch), block 128 (4 waves).
// ---------------------------------------------------------------------------
__global__ void sim_wmma(const _Float16* __restrict__ ip, const _Float16* __restrict__ tp,
                         float* __restrict__ S) {
    __shared__ alignas(16) _Float16 As[2][64 * 32];
    __shared__ alignas(16) _Float16 Bs[2][64 * 32];
    int b  = blockIdx.z;
    int m0 = blockIdx.x * 64;
    int n0 = blockIdx.y * 64;
    int t  = threadIdx.x;
    int wv = t >> 5, lane = t & 31;

    const _Float16* A = ip + (size_t)b * 256 * 256;
    const _Float16* B = tp + (size_t)b * 256 * 256;

    v8f acc[4] = {};
    int arow = t >> 1;
    int aoff = (t & 1) * 16;

    auto stage = [&](int ks, int bufI) {
        int k0 = ks * 32;
        const _Float16* srcA = &A[(size_t)(m0 + arow) * 256 + k0 + aoff];
        g2l_16(srcA,     &As[bufI][arow * 32 + aoff]);
        g2l_16(srcA + 8, &As[bufI][arow * 32 + aoff + 8]);
        const _Float16* srcB = &B[(size_t)(n0 + arow) * 256 + k0 + aoff];
        g2l_16(srcB,     &Bs[bufI][arow * 32 + aoff]);
        g2l_16(srcB + 8, &Bs[bufI][arow * 32 + aoff + 8]);
    };

    stage(0, 0);
    async_fence();
    __syncthreads();

    for (int ks = 0; ks < 8; ++ks) {
        int cur = ks & 1, nxt = cur ^ 1;
        if (ks + 1 < 8) stage(ks + 1, nxt);

        int kb  = (lane >> 4) * 8;
        int kb2 = (lane >> 4) * 16;
        const _Float16* ar = &As[cur][(wv * 16 + (lane & 15)) * 32];
        V16 a;
        a.h[0] = *reinterpret_cast<const v8h*>(ar + kb);
        a.h[1] = *reinterpret_cast<const v8h*>(ar + 16 + kb);
#pragma unroll
        for (int nt = 0; nt < 4; ++nt) {
            const _Float16* br = &Bs[cur][(nt * 16 + (lane & 15)) * 32 + kb];
            V16 bb;
            bb.h[0] = *reinterpret_cast<const v8h*>(br);
            bb.h[1] = *reinterpret_cast<const v8h*>(br + 16);
            acc[nt] = __builtin_amdgcn_wmma_f32_16x16x32_f16(false, a.v, false, bb.v,
                                                             (short)0, acc[nt], false, false);
        }
        async_fence();
        __syncthreads();
    }

#pragma unroll
    for (int nt = 0; nt < 4; ++nt) {
        int n = n0 + nt * 16 + (lane & 15);
        float* dst = &S[((size_t)b * 256 + m0 + wv * 16 + ((lane >> 4) << 3)) * 256 + n];
#pragma unroll
        for (int v = 0; v < 8; ++v)
            dst[(size_t)v * 256] = acc[nt][v] * 10.f;  // 1/TEMP
    }
}

// ---------------------------------------------------------------------------
// Per-row CE: positive logit == diag(S) pre-mask, masked diag is -1e9, so
// loss_row = max + log(sum_q exp(s_pq - max)) - s_pp. Mean via atomicAdd.
// ---------------------------------------------------------------------------
__global__ void zero_out(float* o) { o[0] = 0.f; }

__global__ void row_loss(const float* __restrict__ S, float* __restrict__ out) {
    int p = blockIdx.x, b = blockIdx.y, t = threadIdx.x;
    const float* row = S + ((size_t)b * 256 + p) * 256;
    float s = row[t];

    __shared__ float red[8];
    __shared__ float sm, ssum;

    float m = s;
    for (int o = 16; o > 0; o >>= 1) m = fmaxf(m, __shfl_xor(m, o, 32));
    if ((t & 31) == 0) red[t >> 5] = m;
    __syncthreads();
    if (t == 0) {
        float mm = red[0];
        for (int i = 1; i < 8; ++i) mm = fmaxf(mm, red[i]);
        sm = mm;
    }
    __syncthreads();

    float e = __expf(s - sm);
    for (int o = 16; o > 0; o >>= 1) e += __shfl_xor(e, o, 32);
    if ((t & 31) == 0) red[t >> 5] = e;
    __syncthreads();
    if (t == 0) {
        float ss = 0.f;
        for (int i = 0; i < 8; ++i) ss += red[i];
        ssum = ss;
    }
    __syncthreads();

    if (t == 0) {
        float loss = sm + logf(ssum) - row[p];
        atomicAdd(out, loss * (1.f / (8.f * 256.f)));
    }
}

// ---------------------------------------------------------------------------
// Host-side launcher
// ---------------------------------------------------------------------------
extern "C" void kernel_launch(void* const* d_in, const int* in_sizes, int n_in,
                              void* d_out, int out_size, void* d_ws, size_t ws_size,
                              hipStream_t stream) {
    (void)in_sizes; (void)n_in; (void)out_size; (void)ws_size;

    const float* inp = (const float*)d_in[0];
    const float* tgt = (const float*)d_in[1];
    const float* Wp[7];
    const float* Bp[7];
    for (int l = 0; l < 7; ++l) {
        Wp[l] = (const float*)d_in[2 + 2 * l];
        Bp[l] = (const float*)d_in[3 + 2 * l];
    }
    const int* i_in = (const int*)d_in[16];
    const int* j_in = (const int*)d_in[17];
    const int* i_tg = (const int*)d_in[18];
    const int* j_tg = (const int*)d_in[19];

    // ---- workspace layout (256B aligned) ----
    char* ws = (char*)d_ws;
    size_t off = 0;
    auto alloc = [&](size_t bytes) {
        size_t o = off;
        off = (off + bytes + 255) & ~(size_t)255;
        return ws + o;
    };
    const size_t BUF = (size_t)8 * 224 * 224 * 64 * sizeof(_Float16);  // 51.4 MB
    _Float16* buf0 = (_Float16*)alloc(BUF);
    _Float16* buf1 = (_Float16*)alloc(BUF);

    const int OCs[7] = {64, 64, 128, 128, 256, 256, 256};
    const int ICs[7] = {3, 64, 64, 128, 128, 256, 256};
    const int KPs[7] = {32, 576, 576, 1152, 1152, 2304, 2304};  // conv0 padded 27->32
    _Float16* wts[7];
    for (int l = 0; l < 7; ++l)
        wts[l] = (_Float16*)alloc((size_t)OCs[l] * KPs[l] * sizeof(_Float16));

    float*    ipraw = (float*)alloc((size_t)8 * 256 * 256 * sizeof(float));
    float*    tpraw = (float*)alloc((size_t)8 * 256 * 256 * sizeof(float));
    _Float16* ipf16 = (_Float16*)alloc((size_t)8 * 256 * 256 * sizeof(_Float16));
    _Float16* tpf16 = (_Float16*)alloc((size_t)8 * 256 * 256 * sizeof(_Float16));
    float*    Sbuf  = (float*)alloc((size_t)8 * 256 * 256 * sizeof(float));

    // ---- weight transforms ----
    for (int l = 0; l < 7; ++l) {
        int total = OCs[l] * KPs[l];
        cvt_weights<<<(total + 255) / 256, 256, 0, stream>>>(Wp[l], wts[l],
                                                             OCs[l], ICs[l], KPs[l]);
    }

    // ---- VGG feature chain + patch extraction for one image set ----
    auto run_vgg = [&](const float* img, const int* ii, const int* jj,
                       float* praw, _Float16* pf16) {
        conv0_wmma<<<dim3(784, 1, 8), 128, 0, stream>>>(img, wts[0], Bp[0], buf0);
        conv_wmma<64, 64><<<dim3(392, 1, 8), 128, 0, stream>>>(buf0, wts[1], Bp[1], buf1, 224, 224);
        maxpool2<<<2048, 256, 0, stream>>>(buf1, buf0, 224, 224, 64, 8 * 112 * 112 * 64);
        conv_wmma<64, 128><<<dim3(98, 2, 8), 128, 0, stream>>>(buf0, wts[2], Bp[2], buf1, 112, 112);
        conv_wmma<128, 128><<<dim3(98, 2, 8), 128, 0, stream>>>(buf1, wts[3], Bp[3], buf0, 112, 112);
        maxpool2<<<2048, 256, 0, stream>>>(buf0, buf1, 112, 112, 128, 8 * 56 * 56 * 128);
        conv_wmma<128, 256><<<dim3(25, 4, 8), 128, 0, stream>>>(buf1, wts[4], Bp[4], buf0, 56, 56);
        conv_wmma<256, 256><<<dim3(25, 4, 8), 128, 0, stream>>>(buf0, wts[5], Bp[5], buf1, 56, 56);
        conv_wmma<256, 256><<<dim3(25, 4, 8), 128, 0, stream>>>(buf1, wts[6], Bp[6], buf0, 56, 56);
        patch_mean<<<dim3(256, 8), 256, 0, stream>>>(buf0, ii, jj, praw);
        l2norm_f16<<<dim3(256, 8), 256, 0, stream>>>(praw, pf16);
    };

    run_vgg(inp, i_in, j_in, ipraw, ipf16);
    run_vgg(tgt, i_tg, j_tg, tpraw, tpf16);

    sim_wmma<<<dim3(4, 4, 8), 128, 0, stream>>>(ipf16, tpf16, Sbuf);
    zero_out<<<1, 1, 0, stream>>>((float*)d_out);
    row_loss<<<dim3(256, 8), 256, 0, stream>>>(Sbuf, (float*)d_out);
}
